// DecoderBlock_42666205118547
// MI455X (gfx1250) — compile-verified, matrix operand order checked by reading, not run on hardware
//
#include <hip/hip_runtime.h>
#include <hip/hip_bf16.h>
#include <math.h>

// ---- problem constants (from setup_inputs: B=8, H=W=28, dim=256) ----
#define NB     8
#define H2     56
#define W2     56
#define HW2    3136            // 56*56
#define NP     25088           // 8*3136 pixels after upsample
#define CH     128             // hd
#define MEXP   6272            // 8*28*28 rows of the expand GEMM
#define EPSV   1e-5f

typedef __attribute__((ext_vector_type(16))) __bf16 v16bf;
typedef __attribute__((ext_vector_type(8)))  __bf16 v8bf;
typedef __attribute__((ext_vector_type(8)))  float  v8f;

__device__ __forceinline__ float wave_sum(float v) {
    #pragma unroll
    for (int off = 16; off > 0; off >>= 1) v += __shfl_xor(v, off, 32);
    return v;
}

__device__ __forceinline__ float gelu_exact(float t) {
    return 0.5f * t * (1.0f + erff(t * 0.70710678118654752f));
}

// =====================================================================
// LDS-free WMMA GEMM: C[M,N] = A[M,K] * Bt[N,K]^T, A/Bt bf16, C f32.
// Fragments loaded straight from global per the ISA 16-bit layouts:
//   lane l / l+16 own row (or col) l; K = kb..kb+7 and kb+16..kb+23
//   with kb = 8*(lane>=16)  ->  two aligned b128 loads per fragment.
// Block = 128 threads (4 waves). Tile: 64(M) x 32(N), 2 acc per wave.
// K compile-time -> fully unrolled stream of global_load_b128 + v_wmma.
// Requires M%64==0, N%32==0, K%32==0 (true for all GEMMs here).
// =====================================================================
template<int K>
__global__ void gemm_wmma_direct(const __bf16* __restrict__ A,
                                 const __bf16* __restrict__ Bt,
                                 float* __restrict__ C, int N)
{
    const int lane = threadIdx.x & 31;
    const int wid  = threadIdx.x >> 5;
    const int l    = lane & 15;
    const int half = lane >> 4;
    const int kb   = half * 8;

    const int n0 = blockIdx.x * 32;
    const int m0 = blockIdx.y * 64 + wid * 16;

    const __bf16* ap  = A  + (size_t)(m0 + l) * K + kb;
    const __bf16* bp0 = Bt + (size_t)(n0 + l) * K + kb;
    const __bf16* bp1 = Bt + (size_t)(n0 + 16 + l) * K + kb;

    v8f acc0 = {}, acc1 = {};
    #pragma unroll
    for (int k0 = 0; k0 < K; k0 += 32) {
        union { v16bf v; v8bf h[2]; } a, b0, b1;
        a.h[0]  = *(const v8bf*)(ap  + k0);
        a.h[1]  = *(const v8bf*)(ap  + k0 + 16);
        b0.h[0] = *(const v8bf*)(bp0 + k0);
        b0.h[1] = *(const v8bf*)(bp0 + k0 + 16);
        b1.h[0] = *(const v8bf*)(bp1 + k0);
        b1.h[1] = *(const v8bf*)(bp1 + k0 + 16);
        acc0 = __builtin_amdgcn_wmma_f32_16x16x32_bf16(
                   false, a.v, false, b0.v, (short)0, acc0, false, false);
        acc1 = __builtin_amdgcn_wmma_f32_16x16x32_bf16(
                   false, a.v, false, b1.v, (short)0, acc1, false, false);
    }

    // C layout: VGPR i -> m = i + 8*half ; n = lane&15
    const int mrow = m0 + half * 8;
    #pragma unroll
    for (int i = 0; i < 8; ++i) {
        C[(size_t)(mrow + i) * N + n0 + l]      = acc0[i];
        C[(size_t)(mrow + i) * N + n0 + 16 + l] = acc1[i];
    }
}

// ---- one-time f32 -> bf16 operand conversions ----
__global__ void cvt_bf16(const float* __restrict__ src, __bf16* __restrict__ dst, int n)
{
    const int i = blockIdx.x * 256 + threadIdx.x;
    if (i < n) dst[i] = (__bf16)src[i];
}

// src [rows][cols] f32 -> dst [cols][rows] bf16   (expand_w -> [N,K])
__global__ void transpose_cvt_bf16(const float* __restrict__ src, __bf16* __restrict__ dst,
                                   int rows, int cols)
{
    const int i = blockIdx.x * 256 + threadIdx.x;
    if (i >= rows * cols) return;
    const int r = i / cols, c = i % cols;
    dst[(size_t)c * rows + r] = (__bf16)src[i];
}

// =====================================================================
// Pixel-shuffle of the expand GEMM output + LN(expand) + LN(nd).
// One wave per output pixel. Writes xe_ln (f32, residual), xd (f32, for
// the bilinear gather) and xd_bf (bf16, GEMM A-operand).
// =====================================================================
__global__ void expand_post(const float* __restrict__ xe_raw,
                            const float* __restrict__ eg, const float* __restrict__ eb,
                            const float* __restrict__ ndg, const float* __restrict__ ndb,
                            float* __restrict__ xe_ln, float* __restrict__ xd,
                            __bf16* __restrict__ xd_bf)
{
    const int p    = blockIdx.x * 4 + (threadIdx.x >> 5);
    const int lane = threadIdx.x & 31;
    const int b    = p / HW2, rem = p % HW2;
    const int y    = rem / W2, x = rem % W2;
    const int row  = b * 784 + (y >> 1) * 28 + (x >> 1);
    const int colb = (y & 1) * 256 + (x & 1) * 128;
    const float* src = xe_raw + (size_t)row * 512 + colb;

    float v[4]; float s = 0.f;
    #pragma unroll
    for (int j = 0; j < 4; ++j) { v[j] = src[lane + 32 * j]; s += v[j]; }
    const float mean = wave_sum(s) * (1.0f / 128.0f);
    float q = 0.f;
    #pragma unroll
    for (int j = 0; j < 4; ++j) { float d = v[j] - mean; q += d * d; }
    const float ist = rsqrtf(wave_sum(q) * (1.0f / 128.0f) + EPSV);

    float w[4]; float s2 = 0.f;
    #pragma unroll
    for (int j = 0; j < 4; ++j) {
        const int ch = lane + 32 * j;
        w[j] = (v[j] - mean) * ist * eg[ch] + eb[ch];
        s2 += w[j];
    }
    const float m2 = wave_sum(s2) * (1.0f / 128.0f);
    float q2 = 0.f;
    #pragma unroll
    for (int j = 0; j < 4; ++j) { float d = w[j] - m2; q2 += d * d; }
    const float ist2 = rsqrtf(wave_sum(q2) * (1.0f / 128.0f) + EPSV);

    #pragma unroll
    for (int j = 0; j < 4; ++j) {
        const int ch = lane + 32 * j;
        const float t = (w[j] - m2) * ist2 * ndg[ch] + ndb[ch];
        xe_ln[(size_t)p * CH + ch] = w[j];
        xd[(size_t)p * CH + ch]    = t;
        xd_bf[(size_t)p * CH + ch] = (__bf16)t;
    }
}

// =====================================================================
// 3x3 SAME convs producing 18 offset + 9 sigmoid(mask) channels.
// One lane per (pixel, out-channel); 4 pixels per 128-thread block.
// =====================================================================
__global__ void conv3x3_offmask(const float* __restrict__ toff,
                                const float* __restrict__ tmsk,
                                const float* __restrict__ w_off,   // [18,128,3,3]
                                const float* __restrict__ w_msk,   // [9,128,3,3]
                                float* __restrict__ om)            // [NP,27]
{
    const int p    = blockIdx.x * 4 + (threadIdx.x >> 5);
    const int slot = threadIdx.x & 31;
    if (slot >= 27) return;
    const int b = p / HW2, rem = p % HW2;
    const int y = rem / W2, x = rem % W2;

    const float* src = (slot < 18) ? toff : tmsk;
    const float* wgt = (slot < 18) ? (w_off + (size_t)slot * 1152)
                                   : (w_msk + (size_t)(slot - 18) * 1152);
    float acc = 0.f;
    for (int ky = 0; ky < 3; ++ky) {
        const int yy = y + ky - 1;
        if (yy < 0 || yy >= H2) continue;
        for (int kx = 0; kx < 3; ++kx) {
            const int xx = x + kx - 1;
            if (xx < 0 || xx >= W2) continue;
            const float* t    = src + (size_t)(b * HW2 + yy * W2 + xx) * CH;
            const float* wrow = wgt + (ky * 3 + kx);      // stride 9 over i
            for (int i = 0; i < CH; ++i)
                acc = fmaf(t[i], wrow[(size_t)i * 9], acc);
        }
    }
    if (slot >= 18) acc = 1.0f / (1.0f + expf(-acc));     // sigmoid(mask)
    om[(size_t)p * 27 + slot] = acc;
}

// =====================================================================
// Deformable bilinear gather + mask + depthwise 3x3 combine (+bias).
// One thread per (pixel, channel); offsets are wave-uniform (scalar loads).
// =====================================================================
__global__ void deform_gather(const float* __restrict__ xd,   // [NP,128]
                              const float* __restrict__ om,   // [NP,27]
                              const float* __restrict__ dww,  // [128,9]
                              const float* __restrict__ dwb,  // [128]
                              float* __restrict__ draw)       // [NP,128]
{
    const size_t idx = (size_t)blockIdx.x * 256 + threadIdx.x;
    const int ch = (int)(idx & (CH - 1));
    const int p  = (int)(idx >> 7);
    const int b  = p / HW2, rem = p % HW2;
    const int y  = rem / W2, x = rem % W2;
    const float* o = om + (size_t)p * 27;

    float acc = 0.f;
    #pragma unroll
    for (int kk = 0; kk < 9; ++kk) {
        const float py  = (float)y + (float)(kk / 3 - 1) + o[2 * kk];
        const float px  = (float)x + (float)(kk % 3 - 1) + o[2 * kk + 1];
        const float msk = o[18 + kk];
        const float fy0 = floorf(py), fx0 = floorf(px);
        float v = 0.f;
        #pragma unroll
        for (int t = 0; t < 4; ++t) {
            const float yi = fy0 + (float)(t >> 1);
            const float xi = fx0 + (float)(t & 1);
            const float wy = 1.0f - fabsf(py - yi);
            const float wx = 1.0f - fabsf(px - xi);
            const float valid = (yi >= 0.f && yi <= (float)(H2 - 1) &&
                                 xi >= 0.f && xi <= (float)(W2 - 1)) ? 1.f : 0.f;
            const int qy = (int)fminf(fmaxf(yi, 0.f), (float)(H2 - 1));
            const int qx = (int)fminf(fmaxf(xi, 0.f), (float)(W2 - 1));
            v = fmaf(wy * wx * valid,
                     xd[((size_t)b * HW2 + qy * W2 + qx) * CH + ch], v);
        }
        acc = fmaf(msk * v, dww[ch * 9 + kk], acc);
    }
    draw[idx] = acc + dwb[ch];
}

// =====================================================================
// GroupNorm stats (32 groups of 4 channels over 3136 pixels).
// One 256-thread block per (batch, group): grid = 256.
// stats[bg*2] = mean ; stats[bg*2+1] = rsqrt(var+eps)
// =====================================================================
__global__ void gn_stats(const float* __restrict__ d, float* __restrict__ stats)
{
    __shared__ float ssum[256], ssq[256];
    const int bg = blockIdx.x;
    const int b  = bg >> 5, g = bg & 31;
    const int tid = threadIdx.x;
    float s = 0.f, q = 0.f;
    for (int pix = tid; pix < HW2; pix += 256) {
        const float* row = d + ((size_t)b * HW2 + pix) * CH + g * 4;
        #pragma unroll
        for (int j = 0; j < 4; ++j) { const float v = row[j]; s += v; q = fmaf(v, v, q); }
    }
    ssum[tid] = s; ssq[tid] = q;
    __syncthreads();
    for (int st = 128; st > 0; st >>= 1) {
        if (tid < st) { ssum[tid] += ssum[tid + st]; ssq[tid] += ssq[tid + st]; }
        __syncthreads();
    }
    if (tid == 0) {
        const float n    = (float)(HW2 * 4);
        const float mean = ssum[0] / n;
        const float var  = ssq[0] / n - mean * mean;
        stats[bg * 2]     = mean;
        stats[bg * 2 + 1] = rsqrtf(var + EPSV);
    }
}

// GroupNorm-apply + GELU + LayerNorm(na) -> bf16 GEMM operand.
__global__ void d_post(const float* __restrict__ draw, const float* __restrict__ stats,
                       const float* __restrict__ gng, const float* __restrict__ gnb,
                       const float* __restrict__ nag, const float* __restrict__ nab,
                       __bf16* __restrict__ dln)
{
    const int p    = blockIdx.x * 4 + (threadIdx.x >> 5);
    const int lane = threadIdx.x & 31;
    const int b    = p / HW2;

    float v[4]; float s = 0.f;
    #pragma unroll
    for (int j = 0; j < 4; ++j) {
        const int ch = lane + 32 * j;
        const int g  = ch >> 2;
        const float mean = stats[(b * 32 + g) * 2];
        const float ist  = stats[(b * 32 + g) * 2 + 1];
        float t = (draw[(size_t)p * CH + ch] - mean) * ist * gng[ch] + gnb[ch];
        t = gelu_exact(t);
        v[j] = t; s += t;
    }
    const float m = wave_sum(s) * (1.0f / 128.0f);
    float q = 0.f;
    #pragma unroll
    for (int j = 0; j < 4; ++j) { const float d = v[j] - m; q += d * d; }
    const float ist2 = rsqrtf(wave_sum(q) * (1.0f / 128.0f) + EPSV);
    #pragma unroll
    for (int j = 0; j < 4; ++j) {
        const int ch = lane + 32 * j;
        dln[(size_t)p * CH + ch] = (__bf16)((v[j] - m) * ist2 * nag[ch] + nab[ch]);
    }
}

// Residual add + LN(dec) and LN(skip); writes concat [xs | xdn] as bf16.
__global__ void fuse_cat(const float* __restrict__ xe_ln, const float* __restrict__ proj,
                         const float* __restrict__ xskip,
                         const float* __restrict__ sg, const float* __restrict__ sb,
                         const float* __restrict__ dg, const float* __restrict__ db,
                         __bf16* __restrict__ cat)                // [NP,256] bf16
{
    const int p    = blockIdx.x * 4 + (threadIdx.x >> 5);
    const int lane = threadIdx.x & 31;

    // ---- xdn = LN(xe_ln + proj, dec) ----
    float a[4]; float s = 0.f;
    #pragma unroll
    for (int j = 0; j < 4; ++j) {
        const int ch = lane + 32 * j;
        a[j] = xe_ln[(size_t)p * CH + ch] + proj[(size_t)p * CH + ch];
        s += a[j];
    }
    float m = wave_sum(s) * (1.0f / 128.0f);
    float q = 0.f;
    #pragma unroll
    for (int j = 0; j < 4; ++j) { const float d = a[j] - m; q += d * d; }
    float ist = rsqrtf(wave_sum(q) * (1.0f / 128.0f) + EPSV);
    #pragma unroll
    for (int j = 0; j < 4; ++j) {
        const int ch = lane + 32 * j;
        cat[(size_t)p * 256 + 128 + ch] = (__bf16)((a[j] - m) * ist * dg[ch] + db[ch]);
    }

    // ---- xs = LN(x_skip, skip) ----
    float k[4]; s = 0.f;
    #pragma unroll
    for (int j = 0; j < 4; ++j) {
        const int ch = lane + 32 * j;
        k[j] = xskip[(size_t)p * CH + ch];
        s += k[j];
    }
    m = wave_sum(s) * (1.0f / 128.0f);
    q = 0.f;
    #pragma unroll
    for (int j = 0; j < 4; ++j) { const float d = k[j] - m; q += d * d; }
    ist = rsqrtf(wave_sum(q) * (1.0f / 128.0f) + EPSV);
    #pragma unroll
    for (int j = 0; j < 4; ++j) {
        const int ch = lane + 32 * j;
        cat[(size_t)p * 256 + ch] = (__bf16)((k[j] - m) * ist * sg[ch] + sb[ch]);
    }
}

// Final GroupNorm-apply + GELU -> output (already pixel-major).
__global__ void final_post(const float* __restrict__ f, const float* __restrict__ stats,
                           const float* __restrict__ hg, const float* __restrict__ hb,
                           float* __restrict__ out)
{
    const size_t idx = (size_t)blockIdx.x * 256 + threadIdx.x;
    const int ch = (int)(idx & (CH - 1));
    const int p  = (int)(idx >> 7);
    const int b  = p / HW2;
    const int g  = ch >> 2;
    const float mean = stats[(b * 32 + g) * 2];
    const float ist  = stats[(b * 32 + g) * 2 + 1];
    const float t = (f[idx] - mean) * ist * hg[ch] + hb[ch];
    out[idx] = gelu_exact(t);
}

// =====================================================================
extern "C" void kernel_launch(void* const* d_in, const int* in_sizes, int n_in,
                              void* d_out, int out_size, void* d_ws, size_t ws_size,
                              hipStream_t stream)
{
    const float* x        = (const float*)d_in[0];
    const float* x_skip   = (const float*)d_in[1];
    const float* expand_w = (const float*)d_in[2];
    const float* expand_g = (const float*)d_in[3];
    const float* expand_b = (const float*)d_in[4];
    const float* nd_g     = (const float*)d_in[5];
    const float* nd_b     = (const float*)d_in[6];
    const float* off_w1   = (const float*)d_in[7];
    const float* off_w2   = (const float*)d_in[8];
    const float* msk_w1   = (const float*)d_in[9];
    const float* msk_w2   = (const float*)d_in[10];
    const float* dw_w     = (const float*)d_in[11];
    const float* dw_b     = (const float*)d_in[12];
    const float* gn_g     = (const float*)d_in[13];
    const float* gn_b     = (const float*)d_in[14];
    const float* na_g     = (const float*)d_in[15];
    const float* na_b     = (const float*)d_in[16];
    const float* proj_w   = (const float*)d_in[17];
    const float* skip_g   = (const float*)d_in[18];
    const float* skip_b   = (const float*)d_in[19];
    const float* dec_g    = (const float*)d_in[20];
    const float* dec_b    = (const float*)d_in[21];
    const float* haf_w    = (const float*)d_in[22];
    const float* hgn_g    = (const float*)d_in[23];
    const float* hgn_b    = (const float*)d_in[24];

    // ---- workspace layout ----
    const size_t SZ = (size_t)NP * CH;              // 3,211,264
    float* ws      = (float*)d_ws;
    float* xe_raw  = ws;                            // [6272,512] == SZ floats
    float* xe_ln   = ws + SZ;                       // [NP,128]
    float* xd      = ws + 2 * SZ;                   // [NP,128] (f32 for gather)
    float* toff    = ws + 3 * SZ;                   // [NP,128] -> reused as d_raw
    float* tmsk    = ws + 4 * SZ;                   // [NP,128] -> reused as f_raw
    float* om      = ws + 5 * SZ;                   // [NP,27]
    float* st1     = om + (size_t)NP * 27;          // 512
    float* st2     = st1 + 512;                     // 512
    float* projout = xe_raw;                        // reuse: xe_raw dead
    float* draw    = toff;                          // reuse: t_off dead
    float* fraw    = tmsk;                          // reuse: t_msk dead

    __bf16* bb      = (__bf16*)(st2 + 512);
    __bf16* xbf     = bb;                           // [6272,256]
    __bf16* xd_bf   = xbf    + (size_t)MEXP * 256;  // [NP,128]
    __bf16* dln_bf  = xd_bf  + SZ;                  // [NP,128]
    __bf16* cat_bf  = dln_bf + SZ;                  // [NP,256]
    __bf16* wexp_t  = cat_bf + 2 * SZ;              // [512,256]
    __bf16* w1o_bf  = wexp_t + 512 * 256;           // [128,128]
    __bf16* w1m_bf  = w1o_bf + 128 * 128;           // [128,128]
    __bf16* pj_bf   = w1m_bf + 128 * 128;           // [128,128]
    __bf16* hf_bf   = pj_bf  + 128 * 128;           // [128,256]

    // 0) one-time operand conversions (independent)
    cvt_bf16<<<(MEXP * 256) / 256, 256, 0, stream>>>(x, xbf, MEXP * 256);
    transpose_cvt_bf16<<<(256 * 512) / 256, 256, 0, stream>>>(expand_w, wexp_t, 256, 512);
    cvt_bf16<<<64, 256, 0, stream>>>(off_w1, w1o_bf, 128 * 128);
    cvt_bf16<<<64, 256, 0, stream>>>(msk_w1, w1m_bf, 128 * 128);
    cvt_bf16<<<64, 256, 0, stream>>>(proj_w, pj_bf, 128 * 128);
    cvt_bf16<<<128, 256, 0, stream>>>(haf_w, hf_bf, 128 * 256);

    const dim3 blk(128);

    // 1) expand GEMM: [6272,256] x [512,256]^T
    gemm_wmma_direct<256><<<dim3(512 / 32, MEXP / 64), blk, 0, stream>>>(
        xbf, wexp_t, xe_raw, 512);
    // 2) pixel-shuffle + LN(expand) + LN(nd)
    expand_post<<<NP / 4, 128, 0, stream>>>(xe_raw, expand_g, expand_b,
                                            nd_g, nd_b, xe_ln, xd, xd_bf);
    // 3) 1x1 convs (weights are [o,i] == [N,K])
    gemm_wmma_direct<128><<<dim3(4, NP / 64), blk, 0, stream>>>(xd_bf, w1o_bf, toff, 128);
    gemm_wmma_direct<128><<<dim3(4, NP / 64), blk, 0, stream>>>(xd_bf, w1m_bf, tmsk, 128);
    // 4) 3x3 convs -> 18 offsets + 9 sigmoid masks
    conv3x3_offmask<<<NP / 4, 128, 0, stream>>>(toff, tmsk, off_w2, msk_w2, om);
    // 5) deformable gather + depthwise combine
    deform_gather<<<(NP * CH) / 256, 256, 0, stream>>>(xd, om, dw_w, dw_b, draw);
    // 6) GroupNorm stats, apply + GELU + LN(na) -> bf16
    gn_stats<<<256, 256, 0, stream>>>(draw, st1);
    d_post<<<NP / 4, 128, 0, stream>>>(draw, st1, gn_g, gn_b, na_g, na_b, dln_bf);
    // 7) proj GEMM (d @ proj_w.T)
    gemm_wmma_direct<128><<<dim3(4, NP / 64), blk, 0, stream>>>(dln_bf, pj_bf, projout, 128);
    // 8) residual + LN(dec) ; LN(skip) ; concat -> bf16
    fuse_cat<<<NP / 4, 128, 0, stream>>>(xe_ln, projout, x_skip,
                                         skip_g, skip_b, dec_g, dec_b, cat_bf);
    // 9) haf GEMM: [NP,256] x haf_w[128,256]^T
    gemm_wmma_direct<256><<<dim3(4, NP / 64), blk, 0, stream>>>(cat_bf, hf_bf, fraw, 128);
    // 10) final GroupNorm + GELU -> d_out (pixel-major == (B, H2*W2, c))
    gn_stats<<<256, 256, 0, stream>>>(fraw, st2);
    final_post<<<(NP * CH) / 256, 256, 0, stream>>>(fraw, st2, hgn_g, hgn_b, (float*)d_out);
}
